// TGN_44882408243779
// MI455X (gfx1250) — matrix operand order repaired and checked by reading, not compiled, not run on hardware
//
#include <hip/hip_runtime.h>
#include <hip/hip_bf16.h>

typedef __attribute__((ext_vector_type(16))) _Float16 v16h;
typedef __attribute__((ext_vector_type(8)))  _Float16 v8h;
typedef __attribute__((ext_vector_type(8)))  float    v8f;

#define NN   20000      // nodes in subgraph
#define NE   320000     // edges
#define MEMD 128        // memory dim
#define EMB  256        // embedding dim
#define EDIM 130        // raw edge-attr dim
#define EPAD 160        // edge dim padded to multiple of 32 for WMMA K-loop
#define NH   4
#define HC   64

// ---------------------------------------------------------------------------
// WMMA GEMM: out[M,256] = A[M,K](f16,row-major) @ Wt[256,K](f16,N-major) + bias
// One wave -> one 16x64 output strip (4 accumulators sharing one A fragment).
// K is compile-time so the K-loop unrolls into chains of v_wmma_f32_16x16x32_f16.
// ---------------------------------------------------------------------------
template<int K>
__global__ __launch_bounds__(128)
void k_gemm16(const _Float16* __restrict__ A, const _Float16* __restrict__ Bt,
              const float* __restrict__ bias, float* __restrict__ out, int Mtiles)
{
    const int mt = blockIdx.x * 4 + threadIdx.y;       // wave-uniform
    if (mt >= Mtiles) return;                           // whole wave exits: EXEC stays all-1s
    const int lane = threadIdx.x;
    const int m    = lane & 15;                         // row (A) / col (B,D) within tile
    const int hi   = lane >> 4;                         // lane-half selector
    const int row0 = mt << 4;
    const int colB = blockIdx.y << 6;                   // 64-wide strip

    // A fragment: lanes 0-15 carry K=[k0,k0+16), lanes 16-31 K=[k0+16,k0+32).
    const _Float16* Ap = A + (size_t)(row0 + m) * K + (hi << 4);
    // B fragments from N-major Wt: lanes 0-15 carry K=[k0,k0+16), lanes 16-31 K=[k0+16,k0+32).
    const _Float16* Bp0 = Bt + (size_t)(colB +  0 + m) * K + (hi << 4);
    const _Float16* Bp1 = Bt + (size_t)(colB + 16 + m) * K + (hi << 4);
    const _Float16* Bp2 = Bt + (size_t)(colB + 32 + m) * K + (hi << 4);
    const _Float16* Bp3 = Bt + (size_t)(colB + 48 + m) * K + (hi << 4);

    v8f acc0 = {}, acc1 = {}, acc2 = {}, acc3 = {};
    #pragma unroll
    for (int k0 = 0; k0 < K; k0 += 32) {
        v8h a0 = *(const v8h*)(Ap + k0);
        v8h a1 = *(const v8h*)(Ap + k0 + 8);
        v16h a = __builtin_shufflevector(a0, a1, 0,1,2,3,4,5,6,7,8,9,10,11,12,13,14,15);

        v8h b0 = *(const v8h*)(Bp0 + k0);
        v8h b1 = *(const v8h*)(Bp0 + k0 + 8);
        v16h b = __builtin_shufflevector(b0, b1, 0,1,2,3,4,5,6,7,8,9,10,11,12,13,14,15);
        acc0 = __builtin_amdgcn_wmma_f32_16x16x32_f16(false, a, false, b, (short)0, acc0, false, false);

        b0 = *(const v8h*)(Bp1 + k0);
        b1 = *(const v8h*)(Bp1 + k0 + 8);
        b  = __builtin_shufflevector(b0, b1, 0,1,2,3,4,5,6,7,8,9,10,11,12,13,14,15);
        acc1 = __builtin_amdgcn_wmma_f32_16x16x32_f16(false, a, false, b, (short)0, acc1, false, false);

        b0 = *(const v8h*)(Bp2 + k0);
        b1 = *(const v8h*)(Bp2 + k0 + 8);
        b  = __builtin_shufflevector(b0, b1, 0,1,2,3,4,5,6,7,8,9,10,11,12,13,14,15);
        acc2 = __builtin_amdgcn_wmma_f32_16x16x32_f16(false, a, false, b, (short)0, acc2, false, false);

        b0 = *(const v8h*)(Bp3 + k0);
        b1 = *(const v8h*)(Bp3 + k0 + 8);
        b  = __builtin_shufflevector(b0, b1, 0,1,2,3,4,5,6,7,8,9,10,11,12,13,14,15);
        acc3 = __builtin_amdgcn_wmma_f32_16x16x32_f16(false, a, false, b, (short)0, acc3, false, false);
    }

    // D layout: VGPR r -> row (8*hi + r), col (lane&15)
    float* op = out + (size_t)(row0 + (hi << 3)) * EMB + colB + m;
    #pragma unroll
    for (int t = 0; t < 4; ++t) {
        const float bv = bias ? bias[colB + (t << 4) + m] : 0.0f;
        const v8f acc = (t == 0) ? acc0 : (t == 1) ? acc1 : (t == 2) ? acc2 : acc3;
        #pragma unroll
        for (int r = 0; r < 8; ++r)
            op[(size_t)r * EMB + (t << 4)] = acc[r] + bv;
    }
}

// ---------------------------------------------------------------------------
// Data prep kernels (fp32 -> f16 operand buffers, 16B-vectorized)
// ---------------------------------------------------------------------------
__global__ void k_wt_conv(const float* __restrict__ W, _Float16* __restrict__ Wt,
                          int K, int padK)   // W:[K,256] -> Wt:[256,padK] (N-major, zero pad)
{
    int idx = blockIdx.x * 256 + threadIdx.x;
    if (idx >= 256 * padK) return;
    int n = idx / padK, k = idx % padK;
    Wt[idx] = (k < K) ? (_Float16)W[(size_t)k * 256 + n] : (_Float16)0.0f;
}

__global__ void k_gather_x(const float* __restrict__ tbl, const int* __restrict__ n_id,
                           _Float16* __restrict__ xh)
{
    int idx = blockIdx.x * 256 + threadIdx.x;       // NN*16 chunks of 8 elems
    if (idx >= NN * (MEMD / 8)) return;
    int i = idx >> 4, c = idx & 15;
    const float4* src = (const float4*)(tbl + (size_t)n_id[i] * MEMD + c * 8);
    float4 f0 = src[0], f1 = src[1];
    v8h o = { (_Float16)f0.x, (_Float16)f0.y, (_Float16)f0.z, (_Float16)f0.w,
              (_Float16)f1.x, (_Float16)f1.y, (_Float16)f1.z, (_Float16)f1.w };
    *(v8h*)(xh + (size_t)i * MEMD + c * 8) = o;
}

__global__ void k_edge_cast(const float* __restrict__ ea, _Float16* __restrict__ eh)
{
    int idx = blockIdx.x * 256 + threadIdx.x;       // NE*20 chunks of 8 elems
    if (idx >= NE * (EPAD / 8)) return;
    int e = idx / (EPAD / 8), c = idx % (EPAD / 8);
    int k = c * 8;
    v8h o;
    if (k + 8 <= EDIM) {                            // edge_attr rows only 8B-aligned
        const float2* src = (const float2*)(ea + (size_t)e * EDIM + k);
        float2 f0 = src[0], f1 = src[1], f2 = src[2], f3 = src[3];
        o = (v8h){ (_Float16)f0.x, (_Float16)f0.y, (_Float16)f1.x, (_Float16)f1.y,
                   (_Float16)f2.x, (_Float16)f2.y, (_Float16)f3.x, (_Float16)f3.y };
    } else {
        #pragma unroll
        for (int j = 0; j < 8; ++j)
            o[j] = (k + j < EDIM) ? (_Float16)ea[(size_t)e * EDIM + k + j] : (_Float16)0.0f;
    }
    *(v8h*)(eh + (size_t)e * EPAD + k) = o;
}

__global__ void k_relu_cast(const float* __restrict__ in, _Float16* __restrict__ outh)
{
    int idx = blockIdx.x * 256 + threadIdx.x;       // NN*32 chunks of 8 elems
    if (idx >= NN * (EMB / 8)) return;
    const float4* src = (const float4*)(in + (size_t)idx * 8);
    float4 f0 = src[0], f1 = src[1];
    v8h o = { (_Float16)fmaxf(f0.x, 0.f), (_Float16)fmaxf(f0.y, 0.f),
              (_Float16)fmaxf(f0.z, 0.f), (_Float16)fmaxf(f0.w, 0.f),
              (_Float16)fmaxf(f1.x, 0.f), (_Float16)fmaxf(f1.y, 0.f),
              (_Float16)fmaxf(f1.z, 0.f), (_Float16)fmaxf(f1.w, 0.f) };
    *(v8h*)(outh + (size_t)idx * 8) = o;
}

// ---------------------------------------------------------------------------
// Attention: score + segment-max, exp + segment-sum, normalized aggregation
// ---------------------------------------------------------------------------
__global__ void k_init_ms(float* __restrict__ mb, float* __restrict__ sb)
{
    int i = blockIdx.x * 256 + threadIdx.x;
    if (i < NN * NH) { mb[i] = -__builtin_inff(); sb[i] = 0.0f; }
}

__global__ void k_score(const float* __restrict__ q, const float* __restrict__ kn,
                        const float* __restrict__ ebuf, const int* __restrict__ ei,
                        float* __restrict__ alpha, float* __restrict__ mbuf)
{
    int edge = blockIdx.x * 8 + (threadIdx.x >> 5);     // one wave per edge
    if (edge >= NE) return;
    int lane = threadIdx.x & 31;
    int src = ei[edge];
    int dst = ei[NE + edge];
    const float4* qp = (const float4*)(q    + (size_t)dst  * EMB + lane * 8);
    const float4* kp = (const float4*)(kn   + (size_t)src  * EMB + lane * 8);
    const float4* ep = (const float4*)(ebuf + (size_t)edge * EMB + lane * 8);
    float4 q0 = qp[0], q1 = qp[1], k0 = kp[0], k1 = kp[1], e0 = ep[0], e1 = ep[1];
    float sum = q0.x * (k0.x + e0.x) + q0.y * (k0.y + e0.y)
              + q0.z * (k0.z + e0.z) + q0.w * (k0.w + e0.w)
              + q1.x * (k1.x + e1.x) + q1.y * (k1.y + e1.y)
              + q1.z * (k1.z + e1.z) + q1.w * (k1.w + e1.w);
    #pragma unroll
    for (int off = 1; off < 8; off <<= 1) sum += __shfl_xor(sum, off, 32);
    if ((lane & 7) == 0) {                               // 8 lanes per head reduced
        int h = lane >> 3;
        float al = sum * 0.125f;                         // 1/sqrt(64)
        alpha[(size_t)edge * NH + h] = al;
        float* ma = mbuf + (size_t)dst * NH + h;         // float atomic-max trick
        if (al >= 0.0f) atomicMax((int*)ma, __float_as_int(al));
        else            atomicMin((unsigned int*)ma, __float_as_uint(al));
    }
}

__global__ void k_expsum(float* __restrict__ alpha, const float* __restrict__ mbuf,
                         float* __restrict__ sbuf, const int* __restrict__ ei)
{
    int i = blockIdx.x * 256 + threadIdx.x;              // E*NH threads
    if (i >= NE * NH) return;
    int edge = i >> 2, h = i & 3;
    int dst  = ei[NE + edge];
    float a = __expf(alpha[i] - mbuf[(size_t)dst * NH + h]);
    alpha[i] = a;
    atomicAdd(&sbuf[(size_t)dst * NH + h], a);
}

__global__ void k_aggregate(const float* __restrict__ alpha, const float* __restrict__ sbuf,
                            const float* __restrict__ vn, const float* __restrict__ ebuf,
                            const int* __restrict__ ei, float* __restrict__ agg)
{
    int edge = blockIdx.x;                               // one block per edge
    int f = threadIdx.x;                                 // 256 features
    int h = f >> 6;
    int src = ei[edge];
    int dst = ei[NE + edge];
    float w   = alpha[(size_t)edge * NH + h] / sbuf[(size_t)dst * NH + h];
    float val = w * (vn[(size_t)src * EMB + f] + ebuf[(size_t)edge * EMB + f]);
    atomicAdd(&agg[(size_t)dst * EMB + f], val);
}

// ---------------------------------------------------------------------------
// Host orchestration
// ---------------------------------------------------------------------------
static inline void* bump(char*& p, size_t bytes)
{
    void* r = p;
    p += (bytes + 255) & ~(size_t)255;
    return r;
}

extern "C" void kernel_launch(void* const* d_in, const int* in_sizes, int n_in,
                              void* d_out, int out_size, void* d_ws, size_t ws_size,
                              hipStream_t stream)
{
    const int*   n_id = (const int*)  d_in[0];
    const int*   ei   = (const int*)  d_in[1];   // [2,E]
    const float* eatt = (const float*)d_in[2];   // [E,130]
    const float* mtab = (const float*)d_in[3];   // [1e6,128]
    const float* Wq1 = (const float*)d_in[4];  const float* bq1 = (const float*)d_in[5];
    const float* Wk1 = (const float*)d_in[6];  const float* bk1 = (const float*)d_in[7];
    const float* Wv1 = (const float*)d_in[8];  const float* bv1 = (const float*)d_in[9];
    const float* We1 = (const float*)d_in[10];
    const float* Ws1 = (const float*)d_in[11]; const float* bs1 = (const float*)d_in[12];
    const float* Wq2 = (const float*)d_in[13]; const float* bq2 = (const float*)d_in[14];
    const float* Wk2 = (const float*)d_in[15]; const float* bk2 = (const float*)d_in[16];
    const float* Wv2 = (const float*)d_in[17]; const float* bv2 = (const float*)d_in[18];
    const float* We2 = (const float*)d_in[19];
    const float* Ws2 = (const float*)d_in[20]; const float* bs2 = (const float*)d_in[21];
    float* out = (float*)d_out;                 // [N,256]

    char* p = (char*)d_ws;
    _Float16* xh   = (_Float16*)bump(p, (size_t)NN * MEMD * 2);
    _Float16* hh   = (_Float16*)bump(p, (size_t)NN * EMB  * 2);
    _Float16* eh   = (_Float16*)bump(p, (size_t)NE * EPAD * 2);
    _Float16* tWq1 = (_Float16*)bump(p, 256 * MEMD * 2);
    _Float16* tWk1 = (_Float16*)bump(p, 256 * MEMD * 2);
    _Float16* tWv1 = (_Float16*)bump(p, 256 * MEMD * 2);
    _Float16* tWs1 = (_Float16*)bump(p, 256 * MEMD * 2);
    _Float16* tWe1 = (_Float16*)bump(p, 256 * EPAD * 2);
    _Float16* tWq2 = (_Float16*)bump(p, 256 * EMB * 2);
    _Float16* tWk2 = (_Float16*)bump(p, 256 * EMB * 2);
    _Float16* tWv2 = (_Float16*)bump(p, 256 * EMB * 2);
    _Float16* tWs2 = (_Float16*)bump(p, 256 * EMB * 2);
    _Float16* tWe2 = (_Float16*)bump(p, 256 * EPAD * 2);
    float* q    = (float*)bump(p, (size_t)NN * EMB * 4);
    float* kn   = (float*)bump(p, (size_t)NN * EMB * 4);
    float* vn   = (float*)bump(p, (size_t)NN * EMB * 4);
    float* agg  = (float*)bump(p, (size_t)NN * EMB * 4);   // skip + aggregation (layer1)
    float* ebuf = (float*)bump(p, (size_t)NE * EMB * 4);   // edge features e = edge_attr@We
    float* alp  = (float*)bump(p, (size_t)NE * NH  * 4);
    float* mb   = (float*)bump(p, (size_t)NN * NH  * 4);
    float* sb   = (float*)bump(p, (size_t)NN * NH  * 4);
    (void)ws_size; (void)in_sizes; (void)n_in; (void)out_size;

    // ---- operand prep ----
    k_gather_x <<<(NN * (MEMD / 8) + 255) / 256, 256, 0, stream>>>(mtab, n_id, xh);
    k_edge_cast<<<(NE * (EPAD / 8) + 255) / 256, 256, 0, stream>>>(eatt, eh);
    int wtgK = (256 * MEMD + 255) / 256, wtgE = (256 * EPAD + 255) / 256,
        wtg2 = (256 * EMB + 255) / 256;
    k_wt_conv<<<wtgK, 256, 0, stream>>>(Wq1, tWq1, MEMD, MEMD);
    k_wt_conv<<<wtgK, 256, 0, stream>>>(Wk1, tWk1, MEMD, MEMD);
    k_wt_conv<<<wtgK, 256, 0, stream>>>(Wv1, tWv1, MEMD, MEMD);
    k_wt_conv<<<wtgK, 256, 0, stream>>>(Ws1, tWs1, MEMD, MEMD);
    k_wt_conv<<<wtgE, 256, 0, stream>>>(We1, tWe1, EDIM, EPAD);
    k_wt_conv<<<wtg2, 256, 0, stream>>>(Wq2, tWq2, EMB, EMB);
    k_wt_conv<<<wtg2, 256, 0, stream>>>(Wk2, tWk2, EMB, EMB);
    k_wt_conv<<<wtg2, 256, 0, stream>>>(Wv2, tWv2, EMB, EMB);
    k_wt_conv<<<wtg2, 256, 0, stream>>>(Ws2, tWs2, EMB, EMB);
    k_wt_conv<<<wtgE, 256, 0, stream>>>(We2, tWe2, EDIM, EPAD);

    const int nodeT = NN / 16;            // 1250 row-tiles
    const int edgeT = NE / 16;            // 20000 row-tiles
    dim3 gb(32, 4);
    dim3 gN((nodeT + 3) / 4, EMB / 64);   // 64-wide column strips
    dim3 gE((edgeT + 3) / 4, EMB / 64);

    // ---- layer 1: WMMA GEMMs ----
    k_gemm16<MEMD><<<gN, gb, 0, stream>>>(xh, tWq1, bq1, q,   nodeT);
    k_gemm16<MEMD><<<gN, gb, 0, stream>>>(xh, tWk1, bk1, kn,  nodeT);
    k_gemm16<MEMD><<<gN, gb, 0, stream>>>(xh, tWv1, bv1, vn,  nodeT);
    k_gemm16<MEMD><<<gN, gb, 0, stream>>>(xh, tWs1, bs1, agg, nodeT);  // skip init
    k_gemm16<EPAD><<<gE, gb, 0, stream>>>(eh, tWe1, nullptr, ebuf, edgeT);

    // ---- layer 1: segment softmax + aggregation ----
    k_init_ms  <<<(NN * NH + 255) / 256, 256, 0, stream>>>(mb, sb);
    k_score    <<<(NE + 7) / 8, 256, 0, stream>>>(q, kn, ebuf, ei, alp, mb);
    k_expsum   <<<(NE * NH + 255) / 256, 256, 0, stream>>>(alp, mb, sb, ei);
    k_aggregate<<<NE, 256, 0, stream>>>(alp, sb, vn, ebuf, ei, agg);
    k_relu_cast<<<(NN * (EMB / 8) + 255) / 256, 256, 0, stream>>>(agg, hh);

    // ---- layer 2: WMMA GEMMs ----
    k_gemm16<EMB> <<<gN, gb, 0, stream>>>(hh, tWq2, bq2, q,   nodeT);
    k_gemm16<EMB> <<<gN, gb, 0, stream>>>(hh, tWk2, bk2, kn,  nodeT);
    k_gemm16<EMB> <<<gN, gb, 0, stream>>>(hh, tWv2, bv2, vn,  nodeT);
    k_gemm16<EMB> <<<gN, gb, 0, stream>>>(hh, tWs2, bs2, out, nodeT); // skip -> d_out
    k_gemm16<EPAD><<<gE, gb, 0, stream>>>(eh, tWe2, nullptr, ebuf, edgeT);

    // ---- layer 2: segment softmax + aggregation into d_out ----
    k_init_ms  <<<(NN * NH + 255) / 256, 256, 0, stream>>>(mb, sb);
    k_score    <<<(NE + 7) / 8, 256, 0, stream>>>(q, kn, ebuf, ei, alp, mb);
    k_expsum   <<<(NE * NH + 255) / 256, 256, 0, stream>>>(alp, mb, sb, ei);
    k_aggregate<<<NE, 256, 0, stream>>>(alp, sb, vn, ebuf, ei, out);
}
